// SimpleMultiHeadAttention_9337258901670
// MI455X (gfx1250) — matrix operand order repaired and checked by reading, not compiled, not run on hardware
//
#include <hip/hip_runtime.h>
#include <hip/hip_bf16.h>

// MHA: B=4, S=2048, D=1024, H=16, HD=64. bf16 WMMA (f32 accum) everywhere.

#define Bc 4
#define Sc 2048
#define Dc 1024
#define Hc 16
#define HDc 64

typedef __attribute__((ext_vector_type(16))) __bf16 v16bf;
typedef __attribute__((ext_vector_type(8)))  float  v8f;

union Frag {
    v16bf v;
    __bf16 h[16];
    uint4 u[2];
};

__device__ __forceinline__ v8f wmma_bf16(const Frag& a, const Frag& b, v8f c) {
    return __builtin_amdgcn_wmma_f32_16x16x32_bf16(
        /*neg_a=*/false, a.v, /*neg_b=*/false, b.v,
        /*c_mod=*/(short)0, c, /*reuse_a=*/false, /*reuse_b=*/false);
}

// Async DMA: copy 16 bytes global -> LDS, tracked by ASYNCcnt (no VGPR data).
__device__ __forceinline__ void async_copy_b128(unsigned lds_addr,
                                                const __bf16* gptr) {
    asm volatile("global_load_async_to_lds_b128 %0, %1, off"
                 :: "v"(lds_addr), "v"(gptr) : "memory");
}

// ---------------------------------------------------------------------------
// Kernel 1: convert fp32 weights to bf16, transposed: dst[batch][c][r] = src[batch][r][c]
// ---------------------------------------------------------------------------
__global__ void cvt_transpose_kernel(const float* __restrict__ src,
                                     __bf16* __restrict__ dst,
                                     int rows, int cols) {
    int batch = blockIdx.y;
    size_t n = (size_t)rows * cols;
    size_t idx = (size_t)blockIdx.x * blockDim.x + threadIdx.x;
    if (idx >= n) return;
    int r = (int)(idx / cols);
    int c = (int)(idx % cols);
    dst[(size_t)batch * n + (size_t)c * rows + r] =
        (__bf16)src[(size_t)batch * n + (size_t)r * cols + c];
}

// ---------------------------------------------------------------------------
// Kernel 2: QKV projection. One wave -> 16x64 tile of q/k/v for one (b,h,proj).
// q,k stored [bh][S][64] bf16; v stored transposed [bh][64][S] bf16.
// ---------------------------------------------------------------------------
__global__ __launch_bounds__(256) void qkv_proj_kernel(
    const float* __restrict__ q_in, const float* __restrict__ k_in,
    const float* __restrict__ v_in,
    const __bf16* __restrict__ wqT, const __bf16* __restrict__ wkT,
    const __bf16* __restrict__ wvT,
    __bf16* __restrict__ q_o, __bf16* __restrict__ k_o,
    __bf16* __restrict__ vT_o) {
    const int lane = threadIdx.x & 31;
    const int wave = threadIdx.x >> 5;
    const int lr = lane & 15;        // row/col within 16
    const int lh = lane >> 4;        // half select
    const int proj = blockIdx.z;
    const int bh = blockIdx.y;
    const int b = bh >> 4;
    const int m0 = (blockIdx.x * 8 + wave) * 16;

    const float* X = (proj == 0 ? q_in : proj == 1 ? k_in : v_in) +
                     (size_t)b * Sc * Dc;
    const __bf16* WT = (proj == 0 ? wqT : proj == 1 ? wkT : wvT) +
                       (size_t)(bh & 15) * HDc * Dc;   // [64][1024]

    v8f acc[4] = {v8f{}, v8f{}, v8f{}, v8f{}};

    for (int k0 = 0; k0 < Dc; k0 += 32) {
        // A fragment: rows m0+lr, K runs {k0+lh*8+0..7, k0+16+lh*8+0..7}
        const float* pa = X + (size_t)(m0 + lr) * Dc + k0 + lh * 8;
        float fa[16];
        *(float4*)&fa[0]  = *(const float4*)(pa);
        *(float4*)&fa[4]  = *(const float4*)(pa + 4);
        *(float4*)&fa[8]  = *(const float4*)(pa + 16);
        *(float4*)&fa[12] = *(const float4*)(pa + 20);

        // B fragments: issue all 8 b128 loads up front (one clause)
        Frag bf[4];
        const __bf16* pb = WT + (size_t)lr * Dc + k0 + lh * 8;
#pragma unroll
        for (int nt = 0; nt < 4; ++nt) {
            bf[nt].u[0] = *(const uint4*)(pb + (size_t)nt * 16 * Dc);
            bf[nt].u[1] = *(const uint4*)(pb + (size_t)nt * 16 * Dc + 16);
        }

        Frag a;
#pragma unroll
        for (int i = 0; i < 16; ++i) a.h[i] = (__bf16)fa[i];

#pragma unroll
        for (int nt = 0; nt < 4; ++nt)
            acc[nt] = wmma_bf16(a, bf[nt], acc[nt]);
    }

    // Branchless epilogue: pick destination + strides once (wave-uniform).
    __bf16* dst;
    size_t rs, cs;
    if (proj == 0)      { dst = q_o  + (size_t)bh * Sc * HDc; rs = HDc; cs = 1; }
    else if (proj == 1) { dst = k_o  + (size_t)bh * Sc * HDc; rs = HDc; cs = 1; }
    else                { dst = vT_o + (size_t)bh * HDc * Sc; rs = 1;   cs = Sc; }
    __bf16* base = dst + (size_t)(m0 + lh * 8) * rs + (size_t)lr * cs;

    // C layout: lane holds (M = i + lh*8, N = lr) per accumulator tile
#pragma unroll
    for (int nt = 0; nt < 4; ++nt)
#pragma unroll
        for (int i = 0; i < 8; ++i)
            base[(size_t)i * rs + (size_t)(nt * 16) * cs] = (__bf16)acc[nt][i];
}

// ---------------------------------------------------------------------------
// Kernel 3: flash attention. 8 waves/block share (b,h); K/V tiles are staged
// into LDS once per block via async global->LDS DMA (double-buffered), then
// each wave computes a 16-query-row x HD=64 output tile with bf16 WMMA and
// fp32 online softmax. Key-padding mask; fully-masked tiles skipped.
// ---------------------------------------------------------------------------
__global__ __launch_bounds__(256) void flash_attn_kernel(
    const __bf16* __restrict__ q, const __bf16* __restrict__ k,
    const __bf16* __restrict__ vT, const int* __restrict__ valid_lens,
    __bf16* __restrict__ attn_o) {
    __shared__ __bf16 ldsK[2][64 * 64];   // [key][hd]   8KB x2
    __shared__ __bf16 ldsV[2][64 * 64];   // [hd][key]   8KB x2
    __shared__ __bf16 ldsP[8][16 * 64];   // per-wave P tile, 2KB each

    const int tid = threadIdx.x;
    const int lane = tid & 31;
    const int wave = tid >> 5;
    const int lr = lane & 15;
    const int lh = lane >> 4;
    const int bh = blockIdx.y;
    const int b = bh >> 4;
    const int h = bh & 15;
    const int m0 = (blockIdx.x * 8 + wave) * 16;

    const __bf16* Q = q + (size_t)bh * Sc * HDc;
    const __bf16* K = k + (size_t)bh * Sc * HDc;
    const __bf16* V = vT + (size_t)bh * HDc * Sc;   // [64][S]
    const int vlen = valid_lens[b];                 // uniform per block

    // Q A-fragments, kept resident: kk in {0,1} covers hd 0..31 / 32..63
    Frag aq[2];
#pragma unroll
    for (int kk = 0; kk < 2; ++kk) {
        const __bf16* p = Q + (size_t)(m0 + lr) * HDc + kk * 32 + lh * 16;
        aq[kk].u[0] = *(const uint4*)(p);
        aq[kk].u[1] = *(const uint4*)(p + 8);
    }

    float mrow[8], lrow[8];
#pragma unroll
    for (int i = 0; i < 8; ++i) { mrow[i] = -1e30f; lrow[i] = 0.f; }
    v8f acc[4] = {v8f{}, v8f{}, v8f{}, v8f{}};

    const float scale = 0.125f;   // 1/sqrt(64)
    const int tiles = (vlen + 63) >> 6;
    const int tend = tiles * 64;
    __bf16* lp = ldsP[wave];

    // Per-thread chunk coords for staging (16B chunks, 512 per 8KB tile).
    const int c0row = tid >> 3,        c0off = (tid & 7) * 8;          // chunk tid
    const int c1row = (tid + 256) >> 3, c1off = ((tid + 256) & 7) * 8; // chunk tid+256

    // Stage tile 0 into buffer 0.
    {
        async_copy_b128((unsigned)(uintptr_t)(&ldsK[0][c0row * 64 + c0off]),
                        K + (size_t)c0row * HDc + c0off);
        async_copy_b128((unsigned)(uintptr_t)(&ldsK[0][c1row * 64 + c1off]),
                        K + (size_t)c1row * HDc + c1off);
        async_copy_b128((unsigned)(uintptr_t)(&ldsV[0][c0row * 64 + c0off]),
                        V + (size_t)c0row * Sc + c0off);
        async_copy_b128((unsigned)(uintptr_t)(&ldsV[0][c1row * 64 + c1off]),
                        V + (size_t)c1row * Sc + c1off);
    }

    for (int t0 = 0; t0 < tend; t0 += 64) {
        const int buf = (t0 >> 6) & 1;
        // Barrier A: everyone finished reading the buffer we overwrite next.
        __syncthreads();
        if (t0 + 64 < tend) {
            const int nb = buf ^ 1;
            const int nt0 = t0 + 64;
            async_copy_b128((unsigned)(uintptr_t)(&ldsK[nb][c0row * 64 + c0off]),
                            K + (size_t)(nt0 + c0row) * HDc + c0off);
            async_copy_b128((unsigned)(uintptr_t)(&ldsK[nb][c1row * 64 + c1off]),
                            K + (size_t)(nt0 + c1row) * HDc + c1off);
            async_copy_b128((unsigned)(uintptr_t)(&ldsV[nb][c0row * 64 + c0off]),
                            V + (size_t)c0row * Sc + nt0 + c0off);
            async_copy_b128((unsigned)(uintptr_t)(&ldsV[nb][c1row * 64 + c1off]),
                            V + (size_t)c1row * Sc + nt0 + c1off);
            asm volatile("s_wait_asynccnt 0x4" ::: "memory");
        } else {
            asm volatile("s_wait_asynccnt 0x0" ::: "memory");
        }
        // Barrier B: all waves' copies for tile t0 have landed in LDS.
        __syncthreads();

        const __bf16* lk = ldsK[buf];
        const __bf16* lv = ldsV[buf];

        // ---- scores: S_tile = Q K^T, 4 tiles of 16 keys (K frags from LDS)
        v8f sc[4];
#pragma unroll
        for (int nt = 0; nt < 4; ++nt) {
            const __bf16* pk = lk + (nt * 16 + lr) * 64 + lh * 16;
            Frag bk0, bk1;
            bk0.u[0] = *(const uint4*)(pk);
            bk0.u[1] = *(const uint4*)(pk + 8);
            bk1.u[0] = *(const uint4*)(pk + 32);
            bk1.u[1] = *(const uint4*)(pk + 40);
            v8f z = v8f{};
            z = wmma_bf16(aq[0], bk0, z);
            z = wmma_bf16(aq[1], bk1, z);
            sc[nt] = z;
        }

        // ---- scale + padding mask (key index uniform per lane across rows)
#pragma unroll
        for (int nt = 0; nt < 4; ++nt) {
            bool masked = (t0 + nt * 16 + lr) >= vlen;
#pragma unroll
            for (int i = 0; i < 8; ++i) {
                float s = sc[nt][i] * scale;
                sc[nt][i] = masked ? -1e30f : s;
            }
        }

        // ---- row max over 64 keys (reduce across each 16-lane half)
        float mnew[8];
#pragma unroll
        for (int i = 0; i < 8; ++i)
            mnew[i] = fmaxf(fmaxf(sc[0][i], sc[1][i]), fmaxf(sc[2][i], sc[3][i]));
#pragma unroll
        for (int off = 8; off >= 1; off >>= 1)
#pragma unroll
            for (int i = 0; i < 8; ++i)
                mnew[i] = fmaxf(mnew[i], __shfl_xor(mnew[i], off, 32));
#pragma unroll
        for (int i = 0; i < 8; ++i) mnew[i] = fmaxf(mnew[i], mrow[i]);

        // ---- rescale old accumulators
        float r[8];
#pragma unroll
        for (int i = 0; i < 8; ++i) { r[i] = __expf(mrow[i] - mnew[i]); mrow[i] = mnew[i]; }
#pragma unroll
        for (int nt = 0; nt < 4; ++nt)
#pragma unroll
            for (int i = 0; i < 8; ++i) acc[nt][i] *= r[i];

        // ---- exponentiate, row sums
        float rs[8];
#pragma unroll
        for (int i = 0; i < 8; ++i) rs[i] = 0.f;
#pragma unroll
        for (int nt = 0; nt < 4; ++nt)
#pragma unroll
            for (int i = 0; i < 8; ++i) {
                float p = __expf(sc[nt][i] - mnew[i]);
                sc[nt][i] = p;
                rs[i] += p;
            }
#pragma unroll
        for (int off = 8; off >= 1; off >>= 1)
#pragma unroll
            for (int i = 0; i < 8; ++i) rs[i] += __shfl_xor(rs[i], off, 32);
#pragma unroll
        for (int i = 0; i < 8; ++i) lrow[i] = lrow[i] * r[i] + rs[i];

        // ---- P (C layout f32) -> LDS (row-major bf16 16x64) -> A fragments
#pragma unroll
        for (int nt = 0; nt < 4; ++nt)
#pragma unroll
            for (int i = 0; i < 8; ++i)
                lp[(i + lh * 8) * 64 + nt * 16 + lr] = (__bf16)sc[nt][i];
        asm volatile("s_wait_dscnt 0" ::: "memory");

        // ---- O += P @ V (P from LDS, V frags from LDS)
#pragma unroll
        for (int kk = 0; kk < 2; ++kk) {
            const __bf16* pp = lp + lr * 64 + kk * 32 + lh * 16;
            Frag ap;
            ap.u[0] = *(const uint4*)(pp);
            ap.u[1] = *(const uint4*)(pp + 8);
#pragma unroll
            for (int nt = 0; nt < 4; ++nt) {
                const __bf16* pv = lv + (nt * 16 + lr) * 64 + kk * 32 + lh * 16;
                Frag bv;
                bv.u[0] = *(const uint4*)(pv);
                bv.u[1] = *(const uint4*)(pv + 8);
                acc[nt] = wmma_bf16(ap, bv, acc[nt]);
            }
        }
    }

    // ---- normalize + store head-concat bf16: attn_o[b][row][h*64 + col]
    __bf16* obase = attn_o + ((size_t)b * Sc + m0 + lh * 8) * Dc + h * HDc + lr;
#pragma unroll
    for (int nt = 0; nt < 4; ++nt)
#pragma unroll
        for (int i = 0; i < 8; ++i)
            obase[(size_t)i * Dc + nt * 16] = (__bf16)(acc[nt][i] / lrow[i]);
}

// ---------------------------------------------------------------------------
// Kernel 4: output projection. [B*S,1024] bf16 x WoT[1024][1024] bf16 -> f32
// ---------------------------------------------------------------------------
__global__ __launch_bounds__(256) void out_proj_kernel(
    const __bf16* __restrict__ attn, const __bf16* __restrict__ woT,
    float* __restrict__ out) {
    const int lane = threadIdx.x & 31;
    const int wave = threadIdx.x >> 5;
    const int lr = lane & 15;
    const int lh = lane >> 4;
    const int m0 = (blockIdx.x * 8 + wave) * 16;   // 512 m-tiles over B*S
    const int n0 = blockIdx.y * 64;

    v8f acc[4] = {v8f{}, v8f{}, v8f{}, v8f{}};

    for (int k0 = 0; k0 < Dc; k0 += 32) {
        const __bf16* pa = attn + (size_t)(m0 + lr) * Dc + k0 + lh * 16;
        Frag a;
        a.u[0] = *(const uint4*)(pa);
        a.u[1] = *(const uint4*)(pa + 8);

        Frag bf[4];
        const __bf16* pb = woT + (size_t)(n0 + lr) * Dc + k0 + lh * 16;
#pragma unroll
        for (int nt = 0; nt < 4; ++nt) {
            bf[nt].u[0] = *(const uint4*)(pb + (size_t)nt * 16 * Dc);
            bf[nt].u[1] = *(const uint4*)(pb + (size_t)nt * 16 * Dc + 8);
        }
#pragma unroll
        for (int nt = 0; nt < 4; ++nt)
            acc[nt] = wmma_bf16(a, bf[nt], acc[nt]);
    }

    float* obase = out + (size_t)(m0 + lh * 8) * Dc + n0 + lr;
#pragma unroll
    for (int nt = 0; nt < 4; ++nt)
#pragma unroll
        for (int i = 0; i < 8; ++i)
            obase[(size_t)i * Dc + nt * 16] = acc[nt][i];
}

// ---------------------------------------------------------------------------
extern "C" void kernel_launch(void* const* d_in, const int* in_sizes, int n_in,
                              void* d_out, int out_size, void* d_ws, size_t ws_size,
                              hipStream_t stream) {
    const float* queries = (const float*)d_in[0];
    const float* keys    = (const float*)d_in[1];
    const float* values  = (const float*)d_in[2];
    const int*   vlens   = (const int*)d_in[3];
    const float* Wq      = (const float*)d_in[4];
    const float* Wk      = (const float*)d_in[5];
    const float* Wv      = (const float*)d_in[6];
    const float* Wo      = (const float*)d_in[7];
    float* out = (float*)d_out;

    // Workspace layout (bf16 elements)
    const size_t QKV_E  = (size_t)Bc * Hc * Sc * HDc;  // 8,388,608
    const size_t ATTN_E = (size_t)Bc * Sc * Dc;        // 8,388,608
    const size_t W_E    = (size_t)Hc * Dc * HDc;       // 1,048,576 (== Dc*Dc)
    __bf16* ws      = (__bf16*)d_ws;
    __bf16* q_ws    = ws;
    __bf16* k_ws    = q_ws + QKV_E;
    __bf16* vT_ws   = k_ws + QKV_E;
    __bf16* attn_ws = vT_ws + QKV_E;
    __bf16* wqT     = attn_ws + ATTN_E;
    __bf16* wkT     = wqT + W_E;
    __bf16* wvT     = wkT + W_E;
    __bf16* woT     = wvT + W_E;

    // 1) weight convert + transpose
    cvt_transpose_kernel<<<dim3(256, Hc), 256, 0, stream>>>(Wq, wqT, Dc, HDc);
    cvt_transpose_kernel<<<dim3(256, Hc), 256, 0, stream>>>(Wk, wkT, Dc, HDc);
    cvt_transpose_kernel<<<dim3(256, Hc), 256, 0, stream>>>(Wv, wvT, Dc, HDc);
    cvt_transpose_kernel<<<dim3(4096, 1), 256, 0, stream>>>(Wo, woT, Dc, Dc);

    // 2) QKV projection: 128 m-tiles x 64 (b,h) x 3 projections, 8 waves/block
    qkv_proj_kernel<<<dim3(16, Bc * Hc, 3), 256, 0, stream>>>(
        queries, keys, values, wqT, wkT, wvT, q_ws, k_ws, vT_ws);

    // 3) flash attention: 128 m-tiles x 64 (b,h)
    flash_attn_kernel<<<dim3(16, Bc * Hc), 256, 0, stream>>>(
        q_ws, k_ws, vT_ws, vlens, attn_ws);

    // 4) output projection: 512 m-tiles x 16 n-tiles
    out_proj_kernel<<<dim3(64, 16), 256, 0, stream>>>(attn_ws, woT, out);
}
